// HAGMoE_17265768530005
// MI455X (gfx1250) — matrix operand order repaired and checked by reading, not compiled, not run on hardware
//
#include <hip/hip_runtime.h>
#include <cstdint>
#include <cstddef>

typedef unsigned short u16;
typedef __attribute__((ext_vector_type(16))) __bf16 bf16x16;
typedef __attribute__((ext_vector_type(8)))  float   floatx8;

#define BB 256
#define SS 256
#define TT 16
#define HH 1024
#define NHH 8
#define DD 128
#define GG 3
#define EE 8
#define II 4096
#define RR 256
#define LL 3

#define FLAG_ACC 1
#define FLAG_OBF 2

union ABFrag { uint4 u[2]; bf16x16 v; };

__device__ __forceinline__ u16 f2bf(float x) {
    unsigned u = __float_as_uint(x);
    unsigned r = (u + 0x7FFFu + ((u >> 16) & 1u)) >> 16;
    return (u16)r;
}
__device__ __forceinline__ float bf2f(u16 h) {
    return __uint_as_float(((unsigned)h) << 16);
}

// ---------------------------------------------------------------------------
// Generic bf16 WMMA GEMM:  C[M,N] = A[M,K] (bf16, row-major, lda) * W[N,K]^T
//                                   (+ bias[N])  (+= if FLAG_ACC) (bf16 out if FLAG_OBF)
// grid = (N/64, M/64), block = 128 (4 waves); wave w owns rows [.. +16), cols 64.
// Fragment layouts: 16-bit A 16x32, B 32x16 use contiguous k = 16*half + s in both; f32 C 16x16.
// ---------------------------------------------------------------------------
__global__ __launch_bounds__(128) void gemm_bf16_wmma(
    const u16* __restrict__ A, int lda,
    const u16* __restrict__ W,
    const float* __restrict__ bias,
    void* __restrict__ Cp, int ldc,
    int K, int flags)
{
    const int lane = threadIdx.x & 31;
    const int wave = threadIdx.x >> 5;
    const int half = lane >> 4;      // 0: lanes 0-15, 1: lanes 16-31
    const int l15  = lane & 15;
    const int mbase = (blockIdx.y * 4 + wave) * 16;
    const int nbase = blockIdx.x * 64;

    const floatx8 zero = {0.f,0.f,0.f,0.f,0.f,0.f,0.f,0.f};
    floatx8 c[4] = {zero, zero, zero, zero};

    // A: lane holds row (mbase+l15); 16 consecutive K at k0+half*16
    const u16* arow = A + (size_t)(mbase + l15) * (size_t)lda + half * 16;
    // B: lane holds column (nbase+t*16+l15); 16 consecutive K at k0+half*16
    const u16* wr0 = W + (size_t)(nbase +  0 + l15) * (size_t)K + half * 16;
    const u16* wr1 = W + (size_t)(nbase + 16 + l15) * (size_t)K + half * 16;
    const u16* wr2 = W + (size_t)(nbase + 32 + l15) * (size_t)K + half * 16;
    const u16* wr3 = W + (size_t)(nbase + 48 + l15) * (size_t)K + half * 16;

    for (int k0 = 0; k0 < K; k0 += 32) {
        ABFrag a;
        a.u[0] = *(const uint4*)(arow + k0);
        a.u[1] = *(const uint4*)(arow + k0 + 8);
        ABFrag b0, b1, b2, b3;
        b0.u[0] = *(const uint4*)(wr0 + k0); b0.u[1] = *(const uint4*)(wr0 + k0 + 8);
        b1.u[0] = *(const uint4*)(wr1 + k0); b1.u[1] = *(const uint4*)(wr1 + k0 + 8);
        b2.u[0] = *(const uint4*)(wr2 + k0); b2.u[1] = *(const uint4*)(wr2 + k0 + 8);
        b3.u[0] = *(const uint4*)(wr3 + k0); b3.u[1] = *(const uint4*)(wr3 + k0 + 8);
        c[0] = __builtin_amdgcn_wmma_f32_16x16x32_bf16(false, a.v, false, b0.v, (short)0, c[0], false, false);
        c[1] = __builtin_amdgcn_wmma_f32_16x16x32_bf16(false, a.v, false, b1.v, (short)0, c[1], false, false);
        c[2] = __builtin_amdgcn_wmma_f32_16x16x32_bf16(false, a.v, false, b2.v, (short)0, c[2], false, false);
        c[3] = __builtin_amdgcn_wmma_f32_16x16x32_bf16(false, a.v, false, b3.v, (short)0, c[3], false, false);
    }

    const bool accum = (flags & FLAG_ACC) != 0;
    const bool obf   = (flags & FLAG_OBF) != 0;
    float* Cf = (float*)Cp;
    u16*   Cb = (u16*)Cp;
    #pragma unroll
    for (int t = 0; t < 4; ++t) {
        const int col = nbase + t * 16 + l15;
        const float bv = bias ? bias[col] : 0.0f;
        #pragma unroll
        for (int i = 0; i < 8; ++i) {
            const int row = mbase + half * 8 + i;       // C: VGPR i = row i (+8 for hi lanes)
            const size_t idx = (size_t)row * (size_t)ldc + col;
            float v = c[t][i] + bv;
            if (obf)        Cb[idx] = f2bf(v);
            else if (accum) Cf[idx] += v;
            else            Cf[idx] = v;
        }
    }
}

// ---------------------------------------------------------------------------
// Attention, few queries (T=16) over many keys (S=256): t2s and ca.
// One block per (b, head). Emits mean-over-queries pooled [B,H] directly.
// ---------------------------------------------------------------------------
__global__ __launch_bounds__(256) void attn_smallq_pool(
    const u16* __restrict__ Q,   // [B*T, H] bf16
    const u16* __restrict__ K,   // [B*S, H] bf16
    const u16* __restrict__ V,   // [B*S, H] bf16
    float* __restrict__ pooled)  // [B, H]
{
    __shared__ float Qs[TT * DD];      // 8 KB
    __shared__ float sc[TT * SS];      // 16 KB (reused as [TT][DD] out buffer)
    __shared__ float red[TT][17];
    const int b   = blockIdx.x >> 3;
    const int nh  = blockIdx.x & 7;
    const int tid = threadIdx.x;

    for (int i = tid; i < TT * DD; i += 256) {
        int t = i >> 7, d = i & 127;
        Qs[i] = bf2f(Q[((size_t)(b * TT + t)) * HH + nh * DD + d]);
    }
    __syncthreads();

    {   // scores: thread <-> key s
        const int s = tid;
        const u16* krow = K + ((size_t)(b * SS + s)) * HH + nh * DD;
        float acc[TT];
        #pragma unroll
        for (int t = 0; t < TT; ++t) acc[t] = 0.f;
        for (int d0 = 0; d0 < DD; d0 += 8) {
            uint4 ku = *(const uint4*)(krow + d0);
            const u16* kp = (const u16*)&ku;
            float kf[8];
            #pragma unroll
            for (int j = 0; j < 8; ++j) kf[j] = bf2f(kp[j]);
            #pragma unroll
            for (int t = 0; t < TT; ++t) {
                float a = acc[t];
                #pragma unroll
                for (int j = 0; j < 8; ++j) a += Qs[t * DD + d0 + j] * kf[j];
                acc[t] = a;
            }
        }
        const float scale = 0.088388347648318447f;   // 1/sqrt(128)
        #pragma unroll
        for (int t = 0; t < TT; ++t) sc[t * SS + s] = acc[t] * scale;
    }
    __syncthreads();

    // softmax per row: row t handled by 16 threads
    const int t = tid >> 4, j = tid & 15;
    float m = -3.4e38f;
    for (int s = j; s < SS; s += 16) m = fmaxf(m, sc[t * SS + s]);
    red[t][j] = m;
    __syncthreads();
    if (j == 0) {
        float mm = red[t][0];
        #pragma unroll
        for (int jj = 1; jj < 16; ++jj) mm = fmaxf(mm, red[t][jj]);
        red[t][16] = mm;
    }
    __syncthreads();
    const float rowmax = red[t][16];
    float ssum = 0.f;
    for (int s = j; s < SS; s += 16) {
        float e = expf(sc[t * SS + s] - rowmax);
        sc[t * SS + s] = e;
        ssum += e;
    }
    red[t][j] = ssum;
    __syncthreads();
    if (j == 0) {
        float s2 = 0.f;
        #pragma unroll
        for (int jj = 0; jj < 16; ++jj) s2 += red[t][jj];
        red[t][16] = s2;
    }
    __syncthreads();
    const float inv = 1.0f / red[t][16];
    for (int s = j; s < SS; s += 16) sc[t * SS + s] *= inv;
    __syncthreads();

    // out = P @ V : thread <-> (query t2, 8-wide d chunk)
    const int t2 = tid >> 4;
    const int d0 = (tid & 15) * 8;
    float acc8[8];
    #pragma unroll
    for (int k = 0; k < 8; ++k) acc8[k] = 0.f;
    for (int s = 0; s < SS; ++s) {
        const float p = sc[t2 * SS + s];
        uint4 vu = *(const uint4*)(V + ((size_t)(b * SS + s)) * HH + nh * DD + d0);
        const u16* vp = (const u16*)&vu;
        #pragma unroll
        for (int k = 0; k < 8; ++k) acc8[k] += p * bf2f(vp[k]);
    }
    __syncthreads();
    #pragma unroll
    for (int k = 0; k < 8; ++k) sc[t2 * DD + d0 + k] = acc8[k];
    __syncthreads();
    if (tid < DD) {   // mean over the TT queries (fixed order -> deterministic)
        float ssm = 0.f;
        #pragma unroll
        for (int tt = 0; tt < TT; ++tt) ssm += sc[tt * DD + tid];
        pooled[(size_t)b * HH + nh * DD + tid] = ssm * (1.0f / TT);
    }
}

// ---------------------------------------------------------------------------
// Attention, many queries (S=256) over few keys (T=16): s2t.
// One block per (b, head); thread <-> query. Emits mean-over-queries pooled [B,H].
// ---------------------------------------------------------------------------
__global__ __launch_bounds__(256) void attn_bigq_pool(
    const u16* __restrict__ Q,   // [B*S, H]
    const u16* __restrict__ K,   // [B*T, H]
    const u16* __restrict__ V,   // [B*T, H]
    float* __restrict__ pooled)  // [B, H]
{
    __shared__ float Ks[TT * DD];      // 8 KB
    __shared__ float Vs[TT * DD];      // 8 KB
    __shared__ float rbuf[16 * 256];   // 16 KB
    const int b  = blockIdx.x >> 3;
    const int nh = blockIdx.x & 7;
    const int s  = threadIdx.x;

    for (int i = s; i < TT * DD; i += 256) {
        int r = i >> 7, d = i & 127;
        size_t gi = ((size_t)(b * TT + r)) * HH + nh * DD + d;
        Ks[i] = bf2f(K[gi]);
        Vs[i] = bf2f(V[gi]);
    }
    __syncthreads();

    const u16* qrow = Q + ((size_t)(b * SS + s)) * HH + nh * DD;
    float scr[TT];
    #pragma unroll
    for (int k = 0; k < TT; ++k) scr[k] = 0.f;
    for (int d0 = 0; d0 < DD; d0 += 8) {
        uint4 qu = *(const uint4*)(qrow + d0);
        const u16* qp = (const u16*)&qu;
        float qf[8];
        #pragma unroll
        for (int j = 0; j < 8; ++j) qf[j] = bf2f(qp[j]);
        #pragma unroll
        for (int k = 0; k < TT; ++k) {
            float a = scr[k];
            #pragma unroll
            for (int j = 0; j < 8; ++j) a += qf[j] * Ks[k * DD + d0 + j];
            scr[k] = a;
        }
    }
    const float scale = 0.088388347648318447f;
    float m = -3.4e38f;
    #pragma unroll
    for (int k = 0; k < TT; ++k) { scr[k] *= scale; m = fmaxf(m, scr[k]); }
    float p[TT], sum = 0.f;
    #pragma unroll
    for (int k = 0; k < TT; ++k) { p[k] = expf(scr[k] - m); sum += p[k]; }
    const float inv = 1.0f / sum;
    #pragma unroll
    for (int k = 0; k < TT; ++k) p[k] *= inv;

    for (int c0 = 0; c0 < DD; c0 += 16) {
        #pragma unroll
        for (int dd = 0; dd < 16; ++dd) {
            float o = 0.f;
            #pragma unroll
            for (int k = 0; k < TT; ++k) o += p[k] * Vs[k * DD + c0 + dd];
            rbuf[dd * 256 + s] = o;
        }
        __syncthreads();
        for (int str = 128; str > 0; str >>= 1) {   // fixed-order tree reduce
            if (s < str) {
                #pragma unroll
                for (int dd = 0; dd < 16; ++dd)
                    rbuf[dd * 256 + s] += rbuf[dd * 256 + s + str];
            }
            __syncthreads();
        }
        if (s < 16)
            pooled[(size_t)b * HH + nh * DD + c0 + s] = rbuf[s * 256] * (1.0f / SS);
        __syncthreads();
    }
}

// ---------------------------------------------------------------------------
// LayerNorm over last dim (C), bf16 out.
// ---------------------------------------------------------------------------
__global__ __launch_bounds__(256) void layernorm_row_bf16(
    const float* __restrict__ x, const float* __restrict__ g,
    const float* __restrict__ bta, u16* __restrict__ out, int C)
{
    __shared__ float red[256];
    __shared__ float stat[2];
    const int b = blockIdx.x, tid = threadIdx.x;
    const float* row = x + (size_t)b * C;
    float sm = 0.f;
    for (int i = tid; i < C; i += 256) sm += row[i];
    red[tid] = sm; __syncthreads();
    for (int st = 128; st > 0; st >>= 1) { if (tid < st) red[tid] += red[tid + st]; __syncthreads(); }
    if (tid == 0) stat[0] = red[0] / C;
    __syncthreads();
    const float mu = stat[0];
    float vs = 0.f;
    for (int i = tid; i < C; i += 256) { float d = row[i] - mu; vs += d * d; }
    red[tid] = vs; __syncthreads();
    for (int st = 128; st > 0; st >>= 1) { if (tid < st) red[tid] += red[tid + st]; __syncthreads(); }
    if (tid == 0) stat[1] = rsqrtf(red[0] / C + 1e-5f);
    __syncthreads();
    const float rstd = stat[1];
    u16* orow = out + (size_t)b * C;
    for (int i = tid; i < C; i += 256)
        orow[i] = f2bf((row[i] - mu) * rstd * g[i] + bta[i]);
}

// ---------------------------------------------------------------------------
// top-2-of-3 group routing + expert softmax -> wbe[b, g*E+e] = gp*ep
// launch <<<1,256>>>, thread <-> batch row.
// ---------------------------------------------------------------------------
__global__ __launch_bounds__(256) void routing_kernel(
    const float* __restrict__ fused, const float* __restrict__ cond,
    const float* __restrict__ gr_w, const float* __restrict__ gr_b,
    const float* __restrict__ er_w, const float* __restrict__ er_b,
    float* __restrict__ wbe)
{
    const int b = threadIdx.x;
    const float* f = fused + (size_t)b * HH;
    float gl[GG];
    for (int g = 0; g < GG; ++g) {
        const float* w = gr_w + (size_t)g * HH;
        float a = gr_b[g];
        for (int h = 0; h < HH; ++h) a += f[h] * w[h];
        gl[g] = a;
    }
    const float mx = fmaxf(gl[0], fmaxf(gl[1], gl[2]));
    const float mn = fminf(gl[0], fminf(gl[1], gl[2]));
    const float thr = gl[0] + gl[1] + gl[2] - mx - mn;   // 2nd largest of 3
    float gp[GG], gs = 0.f;
    for (int g = 0; g < GG; ++g) {
        float v = (gl[g] >= thr) ? gl[g] : -1e9f;
        gp[g] = expf(v - mx);
        gs += gp[g];
    }
    for (int g = 0; g < GG; ++g) gp[g] /= gs;

    const float* c = cond + (size_t)b * HH;
    for (int g = 0; g < GG; ++g) {
        float el[EE], em = -3.4e38f;
        for (int e = 0; e < EE; ++e) {
            const float* w = er_w + ((size_t)(g * EE + e)) * HH;
            float a = er_b[g * EE + e];
            for (int h = 0; h < HH; ++h) a += c[h] * w[h];
            el[e] = a; em = fmaxf(em, a);
        }
        float es = 0.f;
        for (int e = 0; e < EE; ++e) { el[e] = expf(el[e] - em); es += el[e]; }
        const float ie = 1.0f / es;
        for (int e = 0; e < EE; ++e)
            wbe[(size_t)b * (GG * EE) + g * EE + e] = gp[g] * el[e] * ie;
    }
}

// --------------------------- small elementwise kernels ----------------------
__global__ void cvt_flat_f2bf(const float* __restrict__ s, u16* __restrict__ d, size_t n) {
    size_t i = (size_t)blockIdx.x * 256 + threadIdx.x;
    if (i < n) d[i] = f2bf(s[i]);
}
__global__ void cvt_strided_f2bf(const float* __restrict__ s, int sld,
                                 u16* __restrict__ d, int dld, int cols) {
    int r = blockIdx.y;
    int c = blockIdx.x * 256 + threadIdx.x;
    if (c < cols) d[(size_t)r * dld + c] = f2bf(s[(size_t)r * sld + c]);
}
__global__ void mulcvt_kernel(const float* __restrict__ a, const float* __restrict__ b,
                              u16* __restrict__ d, size_t n) {
    size_t i = (size_t)blockIdx.x * 256 + threadIdx.x;
    if (i < n) d[i] = f2bf(a[i] * b[i]);
}
__global__ void copy_f32(const float* __restrict__ s, float* __restrict__ d, size_t n) {
    size_t i = (size_t)blockIdx.x * 256 + threadIdx.x;
    if (i < n) d[i] = s[i];
}
__global__ void combine_kernel(const float* __restrict__ fl, const float* __restrict__ gt,
                               const float* __restrict__ bl, float* __restrict__ fused_f,
                               u16* __restrict__ fused_bf, u16* __restrict__ catc_bf) {
    size_t i = (size_t)blockIdx.x * 256 + threadIdx.x;
    if (i >= (size_t)BB * HH) return;
    float gv = 1.0f / (1.0f + expf(-gt[i]));
    float fv = gv * fl[i] + (1.0f - gv) * bl[i];
    fused_f[i] = fv;
    u16 bv = f2bf(fv);
    fused_bf[i] = bv;
    size_t b = i >> 10, h = i & 1023;
    catc_bf[b * (2 * HH) + h] = bv;
}
__global__ void geluscale_kernel(const float* __restrict__ h1, const float* __restrict__ wbe,
                                 int ge, u16* __restrict__ out) {
    size_t i = (size_t)blockIdx.x * 256 + threadIdx.x;
    if (i >= (size_t)BB * II) return;
    size_t b = i >> 12;
    float x = h1[i];
    float gch = 0.5f * x * (1.0f + erff(x * 0.70710678118654752f));   // exact GELU
    out[i] = f2bf(gch * wbe[b * (GG * EE) + ge]);
}
__global__ void moebias_kernel(const float* __restrict__ wbe, const float* __restrict__ e_b2,
                               float* __restrict__ acc) {
    size_t i = (size_t)blockIdx.x * 256 + threadIdx.x;
    if (i >= (size_t)BB * HH) return;
    size_t b = i >> 10, h = i & 1023;
    float s = 0.f;
    for (int ge = 0; ge < GG * EE; ++ge)
        s += wbe[b * (GG * EE) + ge] * e_b2[(size_t)ge * HH + h];
    acc[i] += s;
}
__global__ void cls_kernel(const float* __restrict__ acc, const float* __restrict__ w,
                           const float* __restrict__ bias, float* __restrict__ out) {
    int i = blockIdx.x * 256 + threadIdx.x;
    if (i >= BB * LL) return;
    int b = i / LL, l = i % LL;
    const float* a = acc + (size_t)b * HH;
    const float* wr = w + (size_t)l * HH;
    float s = bias[l];
    for (int h = 0; h < HH; ++h) s += a[h] * wr[h];
    out[i] = s;
}

// ---------------------------------------------------------------------------
extern "C" void kernel_launch(void* const* d_in, const int* in_sizes, int n_in,
                              void* d_out, int out_size, void* d_ws, size_t ws_size,
                              hipStream_t stream)
{
    (void)in_sizes; (void)n_in; (void)out_size; (void)ws_size;
    const float* h_sent   = (const float*)d_in[0];
    const float* h_term   = (const float*)d_in[1];
    const float* ts_in_w  = (const float*)d_in[2];
    const float* ts_out_w = (const float*)d_in[3];
    const float* st_in_w  = (const float*)d_in[4];
    const float* st_out_w = (const float*)d_in[5];
    const float* ca_in_w  = (const float*)d_in[6];
    const float* ca_out_w = (const float*)d_in[7];
    const float* opq_w    = (const float*)d_in[8];
    const float* fuse_w   = (const float*)d_in[9];
    const float* gate_w   = (const float*)d_in[10];
    const float* bs_w     = (const float*)d_in[11];
    const float* bt_w     = (const float*)d_in[12];
    const float* bo_w     = (const float*)d_in[13];
    const float* cond_w   = (const float*)d_in[14];
    const float* gr_w     = (const float*)d_in[15];
    const float* er_w     = (const float*)d_in[16];
    const float* e_w1     = (const float*)d_in[17];
    const float* e_w2     = (const float*)d_in[18];
    const float* cls_w    = (const float*)d_in[19];
    const float* ts_in_b  = (const float*)d_in[20];
    const float* ts_out_b = (const float*)d_in[21];
    const float* st_in_b  = (const float*)d_in[22];
    const float* st_out_b = (const float*)d_in[23];
    const float* ca_in_b  = (const float*)d_in[24];
    const float* ca_out_b = (const float*)d_in[25];
    const float* fuse_b   = (const float*)d_in[26];
    const float* gate_b   = (const float*)d_in[27];
    const float* bs_b     = (const float*)d_in[28];
    const float* bt_b     = (const float*)d_in[29];
    const float* bo_b     = (const float*)d_in[30];
    const float* cond_b   = (const float*)d_in[31];
    const float* gr_b     = (const float*)d_in[32];
    const float* er_b     = (const float*)d_in[33];
    const float* e_b1     = (const float*)d_in[34];
    const float* e_b2     = (const float*)d_in[35];
    const float* cls_b    = (const float*)d_in[36];
    const float* ln_b     = (const float*)d_in[37];
    const float* ln_g     = (const float*)d_in[38];

    char* ws = (char*)d_ws;
    size_t off = 0;
    auto alloc = [&](size_t bytes) -> char* {
        char* p = ws + off;
        off = (off + bytes + 255) & ~(size_t)255;
        return p;
    };
    const size_t BSH = (size_t)BB * SS * HH;   // 67,108,864
    const size_t BTH = (size_t)BB * TT * HH;   // 4,194,304

    u16* hs_bf  = (u16*)alloc(BSH * 2);
    u16* ht_bf  = (u16*)alloc(BTH * 2);
    u16* w_ts   = (u16*)alloc((size_t)3 * HH * HH * 2);
    u16* w_st   = (u16*)alloc((size_t)3 * HH * HH * 2);
    u16* w_ca   = (u16*)alloc((size_t)3 * HH * HH * 2);
    u16* w_opq  = (u16*)alloc((size_t)HH * HH * 2);
    u16* w_tso  = (u16*)alloc((size_t)HH * HH * 2);
    u16* w_sto  = (u16*)alloc((size_t)HH * HH * 2);
    u16* w_cao  = (u16*)alloc((size_t)HH * HH * 2);
    u16* w_fu   = (u16*)alloc((size_t)HH * 3 * HH * 2);
    u16* w_ga   = (u16*)alloc((size_t)HH * 3 * HH * 2);
    u16* w_bsb  = (u16*)alloc((size_t)RR * HH * 2);
    u16* w_btb  = (u16*)alloc((size_t)RR * HH * 2);
    u16* w_bob  = (u16*)alloc((size_t)HH * RR * 2);
    u16* w_cd   = (u16*)alloc((size_t)HH * 2 * HH * 2);
    u16* w_e1   = (u16*)alloc((size_t)II * HH * 2);     // per-expert staging
    u16* w_e2   = (u16*)alloc((size_t)HH * II * 2);     // per-expert staging
    u16* Kbig   = (u16*)alloc(BSH * 2);
    u16* Vbig   = (u16*)alloc(BSH * 2);
    u16* Qst    = (u16*)alloc(BSH * 2);
    u16* Qts    = (u16*)alloc(BTH * 2);
    u16* Kst    = (u16*)alloc(BTH * 2);
    u16* Vst    = (u16*)alloc(BTH * 2);
    u16* oqb    = (u16*)alloc(BTH * 2);
    u16* Qca    = (u16*)alloc(BTH * 2);
    float* t2s_p = (float*)alloc((size_t)BB * HH * 4);
    float* s2t_p = (float*)alloc((size_t)BB * HH * 4);
    float* ca_p  = (float*)alloc((size_t)BB * HH * 4);
    u16* tsp_bf  = (u16*)alloc((size_t)BB * HH * 2);
    u16* stp_bf  = (u16*)alloc((size_t)BB * HH * 2);
    u16* cap_bf  = (u16*)alloc((size_t)BB * HH * 2);
    float* cat3f   = (float*)alloc((size_t)BB * 3 * HH * 4);
    u16*   cat3_bf = (u16*)alloc((size_t)BB * 3 * HH * 2);
    u16*   ln_bf   = (u16*)alloc((size_t)BB * 3 * HH * 2);
    float* fusedlin = (float*)alloc((size_t)BB * HH * 4);
    float* gatelin  = (float*)alloc((size_t)BB * HH * 4);
    float* bsv      = (float*)alloc((size_t)BB * RR * 4);
    float* btv      = (float*)alloc((size_t)BB * RR * 4);
    u16*   prod_bf  = (u16*)alloc((size_t)BB * RR * 2);
    float* bil      = (float*)alloc((size_t)BB * HH * 4);
    float* fused_f  = (float*)alloc((size_t)BB * HH * 4);
    u16*   fused_bf = (u16*)alloc((size_t)BB * HH * 2);
    u16*   catc_bf  = (u16*)alloc((size_t)BB * 2 * HH * 2);
    float* cond_f   = (float*)alloc((size_t)BB * HH * 4);
    float* wbe      = (float*)alloc((size_t)BB * GG * EE * 4);
    float* h1f      = (float*)alloc((size_t)BB * II * 4);
    u16*   h1bf     = (u16*)alloc((size_t)BB * II * 2);
    float* accf     = (float*)alloc((size_t)BB * HH * 4);

    auto cvt = [&](const float* s, u16* d, size_t n) {
        cvt_flat_f2bf<<<dim3((unsigned)((n + 255) / 256)), dim3(256), 0, stream>>>(s, d, n);
    };
    auto gemm = [&](const u16* A, int lda, const u16* W, int K, const float* bias,
                    void* C, int ldc, int M, int N, int flags) {
        dim3 g((unsigned)(N / 64), (unsigned)(M / 64));
        gemm_bf16_wmma<<<g, dim3(128), 0, stream>>>(A, lda, W, bias, C, ldc, K, flags);
    };

    // ---- bf16 conversions of inputs/weights ----
    cvt(h_sent, hs_bf, BSH);
    cvt(h_term, ht_bf, BTH);
    cvt(ts_in_w,  w_ts,  (size_t)3 * HH * HH);
    cvt(st_in_w,  w_st,  (size_t)3 * HH * HH);
    cvt(ca_in_w,  w_ca,  (size_t)3 * HH * HH);
    cvt(opq_w,    w_opq, (size_t)HH * HH);
    cvt(ts_out_w, w_tso, (size_t)HH * HH);
    cvt(st_out_w, w_sto, (size_t)HH * HH);
    cvt(ca_out_w, w_cao, (size_t)HH * HH);
    cvt(fuse_w,   w_fu,  (size_t)HH * 3 * HH);
    cvt(gate_w,   w_ga,  (size_t)HH * 3 * HH);
    cvt(bs_w,     w_bsb, (size_t)RR * HH);
    cvt(bt_w,     w_btb, (size_t)RR * HH);
    cvt(bo_w,     w_bob, (size_t)HH * RR);
    cvt(cond_w,   w_cd,  (size_t)HH * 2 * HH);

    const int MBS = BB * SS, MBT = BB * TT;

    // ---- t2s attention (Q=h_term, K=V=h_sent), pooled over T ----
    gemm(hs_bf, HH, w_ts + (size_t)1 * HH * HH, HH, ts_in_b + HH,     Kbig, HH, MBS, HH, FLAG_OBF);
    gemm(hs_bf, HH, w_ts + (size_t)2 * HH * HH, HH, ts_in_b + 2 * HH, Vbig, HH, MBS, HH, FLAG_OBF);
    gemm(ht_bf, HH, w_ts,                        HH, ts_in_b,          Qts,  HH, MBT, HH, FLAG_OBF);
    attn_smallq_pool<<<dim3(BB * NHH), dim3(256), 0, stream>>>(Qts, Kbig, Vbig, t2s_p);

    // ---- ca attention (Q=h_term@opq, K=V=h_sent), pooled over T ----
    gemm(hs_bf, HH, w_ca + (size_t)1 * HH * HH, HH, ca_in_b + HH,     Kbig, HH, MBS, HH, FLAG_OBF);
    gemm(hs_bf, HH, w_ca + (size_t)2 * HH * HH, HH, ca_in_b + 2 * HH, Vbig, HH, MBS, HH, FLAG_OBF);
    gemm(ht_bf, HH, w_opq, HH, (const float*)nullptr, oqb, HH, MBT, HH, FLAG_OBF);
    gemm(oqb,   HH, w_ca,  HH, ca_in_b,               Qca, HH, MBT, HH, FLAG_OBF);
    attn_smallq_pool<<<dim3(BB * NHH), dim3(256), 0, stream>>>(Qca, Kbig, Vbig, ca_p);

    // ---- s2t attention (Q=h_sent, K=V=h_term), pooled over S ----
    gemm(hs_bf, HH, w_st,                        HH, st_in_b,          Qst, HH, MBS, HH, FLAG_OBF);
    gemm(ht_bf, HH, w_st + (size_t)1 * HH * HH, HH, st_in_b + HH,     Kst, HH, MBT, HH, FLAG_OBF);
    gemm(ht_bf, HH, w_st + (size_t)2 * HH * HH, HH, st_in_b + 2 * HH, Vst, HH, MBT, HH, FLAG_OBF);
    attn_bigq_pool<<<dim3(BB * NHH), dim3(256), 0, stream>>>(Qst, Kst, Vst, s2t_p);

    // ---- out-projections on pooled vectors -> cat3 = [v_sent, v_term, v_op] ----
    cvt(s2t_p, stp_bf, (size_t)BB * HH);
    cvt(t2s_p, tsp_bf, (size_t)BB * HH);
    cvt(ca_p,  cap_bf, (size_t)BB * HH);
    gemm(stp_bf, HH, w_sto, HH, st_out_b, cat3f,          3 * HH, BB, HH, 0);
    gemm(tsp_bf, HH, w_tso, HH, ts_out_b, cat3f + HH,     3 * HH, BB, HH, 0);
    gemm(cap_bf, HH, w_cao, HH, ca_out_b, cat3f + 2 * HH, 3 * HH, BB, HH, 0);

    // ---- fusion head ----
    cvt(cat3f, cat3_bf, (size_t)BB * 3 * HH);
    layernorm_row_bf16<<<dim3(BB), dim3(256), 0, stream>>>(cat3f, ln_g, ln_b, ln_bf, 3 * HH);
    gemm(ln_bf,   3 * HH, w_fu, 3 * HH, fuse_b, fusedlin, HH, BB, HH, 0);
    gemm(cat3_bf, 3 * HH, w_ga, 3 * HH, gate_b, gatelin,  HH, BB, HH, 0);
    gemm(cat3_bf,      3 * HH, w_bsb, HH, bs_b, bsv, RR, BB, RR, 0);   // v_sent @ bs_w^T
    gemm(cat3_bf + HH, 3 * HH, w_btb, HH, bt_b, btv, RR, BB, RR, 0);   // v_term @ bt_w^T
    mulcvt_kernel<<<dim3((BB * RR + 255) / 256), dim3(256), 0, stream>>>(bsv, btv, prod_bf, (size_t)BB * RR);
    gemm(prod_bf, RR, w_bob, RR, bo_b, bil, HH, BB, HH, 0);
    combine_kernel<<<dim3((BB * HH + 255) / 256), dim3(256), 0, stream>>>(
        fusedlin, gatelin, bil, fused_f, fused_bf, catc_bf);
    cvt_strided_f2bf<<<dim3(HH / 256, BB), dim3(256), 0, stream>>>(
        cat3f + HH, 3 * HH, catc_bf + HH, 2 * HH, HH);                 // v_term into catc
    gemm(catc_bf, 2 * HH, w_cd, 2 * HH, cond_b, cond_f, HH, BB, HH, 0);

    // ---- routing ----
    routing_kernel<<<dim3(1), dim3(256), 0, stream>>>(fused_f, cond_f, gr_w, gr_b, er_w, er_b, wbe);

    // ---- MoE: acc = fused + sum_ge wbe * expert(fused) ----
    copy_f32<<<dim3((BB * HH + 255) / 256), dim3(256), 0, stream>>>(fused_f, accf, (size_t)BB * HH);
    for (int ge = 0; ge < GG * EE; ++ge) {
        cvt(e_w1 + (size_t)ge * II * HH, w_e1, (size_t)II * HH);
        gemm(fused_bf, HH, w_e1, HH, e_b1 + (size_t)ge * II, h1f, II, BB, II, 0);
        geluscale_kernel<<<dim3((BB * II + 255) / 256), dim3(256), 0, stream>>>(h1f, wbe, ge, h1bf);
        cvt(e_w2 + (size_t)ge * HH * II, w_e2, (size_t)HH * II);
        gemm(h1bf, II, w_e2, II, (const float*)nullptr, accf, HH, BB, HH, FLAG_ACC);
    }
    moebias_kernel<<<dim3((BB * HH + 255) / 256), dim3(256), 0, stream>>>(wbe, e_b2, accf);

    // ---- classifier ----
    cls_kernel<<<dim3((BB * LL + 255) / 256), dim3(256), 0, stream>>>(accf, cls_w, cls_b, (float*)d_out);
}
